// Attention1D_17008070492807
// MI455X (gfx1250) — matrix operand order repaired and checked by reading, not compiled, not run on hardware
//
#include <hip/hip_runtime.h>
#include <hip/hip_bf16.h>

#define BB      8
#define CC      1024
#define DD      128
#define LLEN    2048
#define LTILES  (LLEN / 16)   // 128 sixteen-wide l tiles (projection)
#define QTILES  (LLEN / 32)   // 64 thirty-two-wide query tiles (attention)

typedef __attribute__((ext_vector_type(16))) __bf16 v16bf;
typedef __attribute__((ext_vector_type(8)))  __bf16 v8bf;
typedef __attribute__((ext_vector_type(8)))  float  v8f;

static __device__ inline v16bf pack16(v8bf lo, v8bf hi) {
  v16bf r;
#pragma unroll
  for (int e = 0; e < 8; ++e) { r[e] = lo[e]; r[8 + e] = hi[e]; }
  return r;
}

// ---------------------------------------------------------------------------
// Kernel 1: 1x1-conv projections  Q = Wq*x+bq, K = Wk*x+bk, V = Wv*x+bv
// Q,K stored [b][l][d] bf16 (A/B-layout friendly for QK^T),
// V stored [b][c][l] bf16 (B-layout friendly for P*V).
// ---------------------------------------------------------------------------
__global__ __launch_bounds__(256) void proj_kernel(
    const float* __restrict__ x,
    const float* __restrict__ Wq, const float* __restrict__ bq,
    const float* __restrict__ Wk, const float* __restrict__ bk,
    const float* __restrict__ Wv, const float* __restrict__ bv,
    __bf16* __restrict__ qbuf, __bf16* __restrict__ kbuf,
    __bf16* __restrict__ vbuf)
{
  __shared__ __align__(16) __bf16 xs[16][CC + 8];   // [l_local][c], 16B-aligned rows

  const int b  = blockIdx.x / LTILES;
  const int l0 = (blockIdx.x % LTILES) * 16;
  const int tid = threadIdx.x;

  { // stage x[b][:, l0:l0+16] into LDS as bf16
    const int l = tid & 15;
    const float* xp = x + (size_t)b * CC * LLEN + l0 + l;
    for (int c = tid >> 4; c < CC; c += 16)
      xs[l][c] = (__bf16)xp[(size_t)c * LLEN];
  }
  __syncthreads();

  const int w = tid >> 5, lane = tid & 31;
  const int n = lane & 15, h = lane >> 4;   // WMMA lane coords

#pragma unroll 1
  for (int t = 0; t < 10; ++t) {
    const int T = w + 8 * t;                // 0..79 output tiles
    const float* W; const float* bias; int o0, kind;
    if (T < 8)       { W = Wq; bias = bq; o0 = T * 16;        kind = 0; }
    else if (T < 16) { W = Wk; bias = bk; o0 = (T - 8) * 16;  kind = 1; }
    else             { W = Wv; bias = bv; o0 = (T - 16) * 16; kind = 2; }

    v8f acc;
#pragma unroll
    for (int e = 0; e < 8; ++e) acc[e] = 0.f;

    const float* wrow = W + (size_t)(o0 + n) * CC;   // A: lane = row o0+n
#pragma unroll 4
    for (int c0 = 0; c0 < CC; c0 += 32) {
      const float4* pa0 = (const float4*)(wrow + c0 + h * 16);
      const float4* pa1 = (const float4*)(wrow + c0 + h * 16 + 8);
      float4 f0 = pa0[0], f1 = pa0[1], g0 = pa1[0], g1 = pa1[1];
      v16bf a;
      a[0]=(__bf16)f0.x; a[1]=(__bf16)f0.y; a[2]=(__bf16)f0.z; a[3]=(__bf16)f0.w;
      a[4]=(__bf16)f1.x; a[5]=(__bf16)f1.y; a[6]=(__bf16)f1.z; a[7]=(__bf16)f1.w;
      a[8]=(__bf16)g0.x; a[9]=(__bf16)g0.y; a[10]=(__bf16)g0.z; a[11]=(__bf16)g0.w;
      a[12]=(__bf16)g1.x; a[13]=(__bf16)g1.y; a[14]=(__bf16)g1.z; a[15]=(__bf16)g1.w;
      const v8bf* pb = (const v8bf*)&xs[n][c0 + h * 16];
      v16bf bm = pack16(pb[0], pb[1]);
      acc = __builtin_amdgcn_wmma_f32_16x16x32_bf16(
                false, a, false, bm, (short)0, acc, false, false);
    }
#pragma unroll
    for (int r = 0; r < 8; ++r) {
      const int o = o0 + r + 8 * h;
      const float val = acc[r] + bias[o];
      if (kind == 0)      qbuf[((size_t)b * LLEN + l0 + n) * DD + o] = (__bf16)val;
      else if (kind == 1) kbuf[((size_t)b * LLEN + l0 + n) * DD + o] = (__bf16)val;
      else                vbuf[((size_t)b * CC + o) * LLEN + l0 + n] = (__bf16)val;
    }
  }
}

// ---------------------------------------------------------------------------
// Kernel 2: flash attention, 32-query blocks.
// Grid: BB*QTILES blocks, 256 threads (8 waves); each wave owns a 128-wide C
// slice and 2 query row-tiles (acc = 16 O-tiles = 128 VGPRs). Per 32-key
// iteration: K j-block staged in double-buffered LDS once per block (kills
// the 8x redundant K reads), 16 QK^T WMMAs, online softmax, P bounced via
// per-wave LDS into A layout, 16 PV WMMAs with each V B-tile reused twice.
// ---------------------------------------------------------------------------
__global__ __launch_bounds__(256) void attn_kernel(
    const float* __restrict__ x, const float* __restrict__ gamma,
    const __bf16* __restrict__ qbuf, const __bf16* __restrict__ kbuf,
    const __bf16* __restrict__ vbuf, float* __restrict__ out)
{
  __shared__ __align__(16) __bf16 ks[2][32][DD + 8];      // K j-block, double buffered
  __shared__ __align__(16) __bf16 p_lds[8][2][16 * 40];   // per-wave, per-i-tile P

  const int b  = blockIdx.x / QTILES;
  const int i0 = (blockIdx.x % QTILES) * 32;
  const int tid = threadIdx.x;
  const int w = tid >> 5, lane = tid & 31;
  const int n = lane & 15, h = lane >> 4;
  const int c0 = w * 128;

  const __bf16* kg    = kbuf + (size_t)b * LLEN * DD;
  const __bf16* vbase = vbuf + (size_t)b * CC * LLEN;
  // cooperative K staging coords: 8 threads per 128-d row, 32B each
  const int sj = tid >> 3;           // 0..31 key row
  const int sd = (tid & 7) * 16;     // 0..112 d offset (16 halfs = 32B)

  // Q in A layout: 2 i-tiles x 4 d-chunks
  v16bf qa[2][4];
#pragma unroll
  for (int it = 0; it < 2; ++it) {
    const __bf16* qrow = qbuf + ((size_t)b * LLEN + i0 + it * 16 + n) * DD;
#pragma unroll
    for (int t = 0; t < 4; ++t) {
      v8bf s0 = *(const v8bf*)(qrow + t * 32 + h * 16);
      v8bf s1 = *(const v8bf*)(qrow + t * 32 + h * 16 + 8);
      qa[it][t] = pack16(s0, s1);
    }
  }

  float run_m[2][8], run_l[2][8];
  v8f acc[2][8];
#pragma unroll
  for (int it = 0; it < 2; ++it) {
#pragma unroll
    for (int r = 0; r < 8; ++r) { run_m[it][r] = -__builtin_inff(); run_l[it][r] = 0.f; }
#pragma unroll
    for (int ct = 0; ct < 8; ++ct)
#pragma unroll
      for (int e = 0; e < 8; ++e) acc[it][ct][e] = 0.f;
  }

#pragma unroll 1
  for (int j0 = 0; j0 < LLEN; j0 += 32) {
    const int buf = (j0 >> 5) & 1;
    { // stage K[j0:j0+32][0:128] -> LDS (each thread: 32 contiguous bytes)
      const __bf16* src = kg + (size_t)(j0 + sj) * DD + sd;
      if (j0 + 32 < LLEN) __builtin_prefetch(src + 32 * DD, 0, 1);
      v8bf k0 = ((const v8bf*)src)[0], k1 = ((const v8bf*)src)[1];
      v8bf* dst = (v8bf*)&ks[buf][sj][sd];
      dst[0] = k0; dst[1] = k1;
    }
    __syncthreads();   // double buffer: next write to this buf is 2 barriers away

    // ---- QK^T: e[it][jt], K B-tiles from LDS
    v8f e0[2], e1[2];
#pragma unroll
    for (int it = 0; it < 2; ++it)
#pragma unroll
      for (int e = 0; e < 8; ++e) { e0[it][e] = 0.f; e1[it][e] = 0.f; }
#pragma unroll
    for (int t = 0; t < 4; ++t) {
      const v8bf* kp0 = (const v8bf*)&ks[buf][n][t * 32 + h * 16];
      v16bf kb0 = pack16(kp0[0], kp0[1]);
      const v8bf* kp1 = (const v8bf*)&ks[buf][16 + n][t * 32 + h * 16];
      v16bf kb1 = pack16(kp1[0], kp1[1]);
#pragma unroll
      for (int it = 0; it < 2; ++it) {
        e0[it] = __builtin_amdgcn_wmma_f32_16x16x32_bf16(
                     false, qa[it][t], false, kb0, (short)0, e0[it], false, false);
        e1[it] = __builtin_amdgcn_wmma_f32_16x16x32_bf16(
                     false, qa[it][t], false, kb1, (short)0, e1[it], false, false);
      }
    }

    // ---- online softmax per i-tile; row m = r + 8h lives in one 16-lane half
#pragma unroll
    for (int it = 0; it < 2; ++it) {
      __bf16* pw = &p_lds[w][it][0];
#pragma unroll
      for (int r = 0; r < 8; ++r) {
        float m = fmaxf(e0[it][r], e1[it][r]);
        m = fmaxf(m, __shfl_xor(m, 1, 32));
        m = fmaxf(m, __shfl_xor(m, 2, 32));
        m = fmaxf(m, __shfl_xor(m, 4, 32));
        m = fmaxf(m, __shfl_xor(m, 8, 32));
        const float nm = fmaxf(run_m[it][r], m);
        const float sc = __expf(run_m[it][r] - nm);
        const float p0 = __expf(e0[it][r] - nm);
        const float p1 = __expf(e1[it][r] - nm);
        float srow = p0 + p1;
        srow += __shfl_xor(srow, 1, 32);
        srow += __shfl_xor(srow, 2, 32);
        srow += __shfl_xor(srow, 4, 32);
        srow += __shfl_xor(srow, 8, 32);
        run_l[it][r] = run_l[it][r] * sc + srow;
        run_m[it][r] = nm;
#pragma unroll
        for (int ct = 0; ct < 8; ++ct) acc[it][ct][r] *= sc;
        const int irow = r + 8 * h;
        pw[irow * 40 + n]      = (__bf16)p0;
        pw[irow * 40 + 16 + n] = (__bf16)p1;
      }
    }
    asm volatile("s_wait_dscnt 0" ::: "memory");  // same-wave LDS RAW

    // P in A layout, one tile per i-tile
    v16bf pa[2];
#pragma unroll
    for (int it = 0; it < 2; ++it) {
      const __bf16* pw = &p_lds[w][it][0];
      pa[it] = pack16(*(const v8bf*)&pw[n * 40 + h * 16],
                      *(const v8bf*)&pw[n * 40 + h * 16 + 8]);
    }

    // ---- PV: each V B-tile feeds both i-tiles
#pragma unroll
    for (int ct = 0; ct < 8; ++ct) {
      const __bf16* vrow = vbase + (size_t)(c0 + ct * 16 + n) * LLEN + j0 + h * 16;
      if (j0 + 32 < LLEN) __builtin_prefetch(vrow + 32, 0, 1);
      v16bf vb = pack16(((const v8bf*)vrow)[0], ((const v8bf*)vrow)[1]);
      acc[0][ct] = __builtin_amdgcn_wmma_f32_16x16x32_bf16(
                       false, pa[0], false, vb, (short)0, acc[0][ct], false, false);
      acc[1][ct] = __builtin_amdgcn_wmma_f32_16x16x32_bf16(
                       false, pa[1], false, vb, (short)0, acc[1][ct], false, false);
    }
  }

  // ---- finalize: out[b][c][i] = gamma * O[i,c]/l[i] + x[b][c][i]
  const float g = gamma[0];
#pragma unroll
  for (int it = 0; it < 2; ++it)
#pragma unroll
    for (int ct = 0; ct < 8; ++ct) {
      const int c = c0 + ct * 16 + n;
#pragma unroll
      for (int r = 0; r < 8; ++r) {
        const int i = i0 + it * 16 + r + 8 * h;
        const size_t idx = ((size_t)b * CC + c) * LLEN + i;
        out[idx] = g * (acc[it][ct][r] / run_l[it][r]) + x[idx];
      }
    }
}

// ---------------------------------------------------------------------------
extern "C" void kernel_launch(void* const* d_in, const int* in_sizes, int n_in,
                              void* d_out, int out_size, void* d_ws, size_t ws_size,
                              hipStream_t stream)
{
  const float* x     = (const float*)d_in[0];
  const float* Wq    = (const float*)d_in[1];
  const float* bq    = (const float*)d_in[2];
  const float* Wk    = (const float*)d_in[3];
  const float* bk    = (const float*)d_in[4];
  const float* Wv    = (const float*)d_in[5];
  const float* bv    = (const float*)d_in[6];
  const float* gamma = (const float*)d_in[7];
  float* out = (float*)d_out;

  char* ws = (char*)d_ws;
  const size_t qk_bytes = (size_t)BB * LLEN * DD * sizeof(__bf16);  // 4 MB each
  __bf16* qbuf = (__bf16*)(ws);
  __bf16* kbuf = (__bf16*)(ws + qk_bytes);
  __bf16* vbuf = (__bf16*)(ws + 2 * qk_bytes);                      // 32 MB

  proj_kernel<<<dim3(BB * LTILES), dim3(256), 0, stream>>>(
      x, Wq, bq, Wk, bk, Wv, bv, qbuf, kbuf, vbuf);
  attn_kernel<<<dim3(BB * QTILES), dim3(256), 0, stream>>>(
      x, gamma, qbuf, kbuf, vbuf, out);
}
